// GaussianPolicy_89103391522967
// MI455X (gfx1250) — matrix operand order repaired and checked by reading, not compiled, not run on hardware
//
#include <hip/hip_runtime.h>

// GraphNet GaussianPolicy for MI455X (gfx1250, wave32, WMMA).
// v3: each wave computes a 32x64 macro-tile (2 row-tiles x 4 col-tiles,
// 8 f32 accumulators) so each A/B fragment feeds 4x/2x more WMMAs.
// All GEMMs via v_wmma_f32_16x16x32_f16 (f32 accumulate); weights are
// pre-transposed f16 (K zero-padded); segment means fused as
// global_atomic_add_f32 into L2-resident accumulators.

typedef __attribute__((ext_vector_type(16))) _Float16 v16h;
typedef __attribute__((ext_vector_type(8)))  _Float16 v8h;
typedef __attribute__((ext_vector_type(8)))  float    v8f;

#define N_NODES 16384
#define N_EDGES 131072
#define NGRAPH  256

// ---- WMMA f16 fragment layout (CDNA5 ISA 7.12.2, wave32) ----
// A 16x32: lane&15 = row M, h=lane>>4. a[0..7]=A[row][k0+8h..+7],
//          a[8..15]=A[row][k0+16+8h..+7] (two contiguous K-runs).
// B 32x16: lane&15 = col N. b[0..7]=B[k0+8h..+7][col], b[8..15]=B[k0+16+8h..+7][col] (same K order as A).
// C/D f32: c[r] -> D[rowtile + r + 8h][col].

__device__ __forceinline__ void cvt8(v16h& a, int base, float4 u, float4 v) {
  a[base + 0] = (_Float16)u.x; a[base + 1] = (_Float16)u.y;
  a[base + 2] = (_Float16)u.z; a[base + 3] = (_Float16)u.w;
  a[base + 4] = (_Float16)v.x; a[base + 5] = (_Float16)v.y;
  a[base + 6] = (_Float16)v.z; a[base + 7] = (_Float16)v.w;
}

__device__ __forceinline__ void cvt8s(v16h& a, int base, float4 u, float4 v, float s) {
  a[base + 0] = (_Float16)(u.x * s); a[base + 1] = (_Float16)(u.y * s);
  a[base + 2] = (_Float16)(u.z * s); a[base + 3] = (_Float16)(u.w * s);
  a[base + 4] = (_Float16)(v.x * s); a[base + 5] = (_Float16)(v.y * s);
  a[base + 6] = (_Float16)(v.z * s); a[base + 7] = (_Float16)(v.w * s);
}

__device__ __forceinline__ v16h load_a_f32v(const float* A, int lda, int row, int h, int k0) {
  const float* p = A + (size_t)row * lda + k0 + 8 * h;
  float4 x0 = *(const float4*)(p);
  float4 x1 = *(const float4*)(p + 4);
  float4 y0 = *(const float4*)(p + 16);
  float4 y1 = *(const float4*)(p + 20);
  v16h a;
  cvt8(a, 0, x0, x1);
  cvt8(a, 8, y0, y1);
  return a;
}

__device__ __forceinline__ v16h load_a_f32sv(const float* A, int lda, int row, int h,
                                             int k0, float s) {
  const float* p = A + (size_t)row * lda + k0 + 8 * h;
  float4 x0 = *(const float4*)(p);
  float4 x1 = *(const float4*)(p + 4);
  float4 y0 = *(const float4*)(p + 16);
  float4 y1 = *(const float4*)(p + 20);
  v16h a;
  cvt8s(a, 0, x0, x1, s);
  cvt8s(a, 8, y0, y1, s);
  return a;
}

__device__ __forceinline__ v16h load_a_f16v(const _Float16* A, int lda, int row, int h, int k0) {
  const _Float16* p = A + (size_t)row * lda + k0 + 8 * h;
  v8h r0 = *(const v8h*)(p);
  v8h r1 = *(const v8h*)(p + 16);
  v16h a;
#pragma unroll
  for (int e = 0; e < 8; ++e) { a[e] = r0[e]; a[8 + e] = r1[e]; }
  return a;
}

// gathered 16-wide global-feature vector, zero-padded to K=32
__device__ __forceinline__ v16h load_a_g16v(const float* G, int vec, int h) {
  const float* p = G + (size_t)vec * 16 + 8 * h;
  float4 x0 = *(const float4*)(p);
  float4 x1 = *(const float4*)(p + 4);
  v16h a;
  cvt8(a, 0, x0, x1);
#pragma unroll
  for (int e = 8; e < 16; ++e) a[e] = (_Float16)0.f;
  return a;
}

__device__ __forceinline__ v16h load_b_f16T(const _Float16* WT, int Kp, int k0, int col, int h) {
  const _Float16* p = WT + (size_t)col * Kp + k0 + 8 * h;
  v8h r0 = *(const v8h*)(p);
  v8h r1 = *(const v8h*)(p + 16);
  v16h b;
#pragma unroll
  for (int e = 0; e < 8; ++e) { b[e] = r0[e]; b[8 + e] = r1[e]; }
  return b;
}

__device__ __forceinline__ v8f wmma_f16(v16h a, v16h b, v8f c) {
  return __builtin_amdgcn_wmma_f32_16x16x32_f16(false, a, false, b, (short)0, c,
                                                false, false);
}

// ---------------- utility kernels ----------------
__global__ void k_zero(float* p, int n) {
  int i = blockIdx.x * blockDim.x + threadIdx.x;
  if (i < n) p[i] = 0.0f;
}

// transpose + convert weights: W[K x N] f32 -> WT[N x Kp] f16, zero K-pad
__global__ void k_wprep(const float* __restrict__ W, _Float16* __restrict__ WT,
                        int K, int N, int Kp) {
  int i = blockIdx.x * blockDim.x + threadIdx.x;
  if (i < N * Kp) {
    int n = i / Kp, k = i - n * Kp;
    WT[i] = (k < K) ? (_Float16)W[(size_t)k * N + n] : (_Float16)0.f;
  }
}

__global__ void k_count_edges(const int* recv, const int* eg, float* cnt_node,
                              float* cnt_eg) {
  int i = blockIdx.x * blockDim.x + threadIdx.x;
  if (i < N_EDGES) {
    atomicAdd(&cnt_node[recv[i]], 1.0f);
    atomicAdd(&cnt_eg[eg[i]], 1.0f);
  }
}

__global__ void k_count_nodes(const int* ng, float* cnt_ng) {
  int i = blockIdx.x * blockDim.x + threadIdx.x;
  if (i < N_NODES) atomicAdd(&cnt_ng[ng[i]], 1.0f);
}

// ---------------- stage 1: e1 = relu(E @ We1 + be1), scatter to node sums ---
// wave macro-tile: 32 rows x 64 cols; N=256 -> 4 col-groups; M -> 4096 pairs
__global__ void k_edge1(const float* __restrict__ EFm, const _Float16* __restrict__ We1T,
                        const float* __restrict__ be1, const int* __restrict__ recv,
                        _Float16* __restrict__ e1h, float* __restrict__ nsum1) {
  int wv = blockIdx.x * 8 + (threadIdx.x >> 5);
  int lane = threadIdx.x & 31, h = lane >> 4, l = lane & 15;
  int mp = wv >> 2, ngp = wv & 3;
  int row0 = mp * 32, colb = ngp * 64;
  v8f acc[2][4] = {};
  v16h a0 = load_a_f32v(EFm, 32, row0 + l, h, 0);
  v16h a1 = load_a_f32v(EFm, 32, row0 + 16 + l, h, 0);
#pragma unroll
  for (int j = 0; j < 4; ++j) {
    v16h b = load_b_f16T(We1T, 32, 0, colb + 16 * j + l, h);
    acc[0][j] = wmma_f16(a0, b, acc[0][j]);
    acc[1][j] = wmma_f16(a1, b, acc[1][j]);
  }
  float bias[4];
#pragma unroll
  for (int j = 0; j < 4; ++j) bias[j] = be1[colb + 16 * j + l];
#pragma unroll
  for (int i = 0; i < 2; ++i) {
#pragma unroll
    for (int r = 0; r < 8; ++r) {
      int rd = row0 + 16 * i + 8 * h + r;
      size_t rbase = (size_t)rd * 256;
      size_t abase = (size_t)recv[rd] * 256;
#pragma unroll
      for (int j = 0; j < 4; ++j) {
        int col = colb + 16 * j + l;
        float v = acc[i][j][r] + bias[j];
        v = v > 0.f ? v : 0.f;
        e1h[rbase + col] = (_Float16)v;
        atomicAdd(&nsum1[abase + col], v);
      }
    }
  }
}

// ---------------- stage 2: n1 = relu(N@Wn1 + mean(e1)@Win1 + bn1) -----------
__global__ void k_node1(const float* __restrict__ NFm, const float* __restrict__ nsum1,
                        const float* __restrict__ cnt_node,
                        const _Float16* __restrict__ Wn1T,
                        const _Float16* __restrict__ Win1T,
                        const float* __restrict__ bn1, _Float16* __restrict__ n1h) {
  int wv = blockIdx.x * 8 + (threadIdx.x >> 5);
  int lane = threadIdx.x & 31, h = lane >> 4, l = lane & 15;
  int mp = wv >> 2, ngp = wv & 3;                 // 512 x 4
  int row0 = mp * 32, colb = ngp * 64;
  v8f acc[2][4] = {};
#pragma unroll
  for (int k0 = 0; k0 < 64; k0 += 32) {
    v16h a0 = load_a_f32v(NFm, 64, row0 + l, h, k0);
    v16h a1 = load_a_f32v(NFm, 64, row0 + 16 + l, h, k0);
#pragma unroll
    for (int j = 0; j < 4; ++j) {
      v16h b = load_b_f16T(Wn1T, 64, k0, colb + 16 * j + l, h);
      acc[0][j] = wmma_f16(a0, b, acc[0][j]);
      acc[1][j] = wmma_f16(a1, b, acc[1][j]);
    }
  }
  float ri0 = 1.0f / fmaxf(cnt_node[row0 + l], 1.0f);
  float ri1 = 1.0f / fmaxf(cnt_node[row0 + 16 + l], 1.0f);
#pragma unroll
  for (int k0 = 0; k0 < 256; k0 += 32) {
    v16h a0 = load_a_f32sv(nsum1, 256, row0 + l, h, k0, ri0);
    v16h a1 = load_a_f32sv(nsum1, 256, row0 + 16 + l, h, k0, ri1);
#pragma unroll
    for (int j = 0; j < 4; ++j) {
      v16h b = load_b_f16T(Win1T, 256, k0, colb + 16 * j + l, h);
      acc[0][j] = wmma_f16(a0, b, acc[0][j]);
      acc[1][j] = wmma_f16(a1, b, acc[1][j]);
    }
  }
  float bias[4];
#pragma unroll
  for (int j = 0; j < 4; ++j) bias[j] = bn1[colb + 16 * j + l];
#pragma unroll
  for (int i = 0; i < 2; ++i) {
#pragma unroll
    for (int r = 0; r < 8; ++r) {
      int rd = row0 + 16 * i + 8 * h + r;
      size_t rbase = (size_t)rd * 256;
#pragma unroll
      for (int j = 0; j < 4; ++j) {
        float v = acc[i][j][r] + bias[j];
        n1h[rbase + colb + 16 * j + l] = (_Float16)(v > 0.f ? v : 0.f);
      }
    }
  }
}

// -------- stage 3: e2 = relu(e1@We2 + G[eg]@Weg2 + be2); never stored, ------
// -------- scattered straight from WMMA accumulators to node & graph sums ----
__global__ void k_edge2(const _Float16* __restrict__ e1h, const float* __restrict__ GFm,
                        const int* __restrict__ eg, const int* __restrict__ recv,
                        const _Float16* __restrict__ We2T,
                        const _Float16* __restrict__ Weg2T,
                        const float* __restrict__ be2, float* __restrict__ nsum2,
                        float* __restrict__ gsum_e) {
  int wv = blockIdx.x * 8 + (threadIdx.x >> 5);
  int lane = threadIdx.x & 31, h = lane >> 4, l = lane & 15;
  int mp = wv >> 1, ngp = wv & 1;                 // 4096 x 2
  int row0 = mp * 32, colb = ngp * 64;
  v8f acc[2][4] = {};
#pragma unroll
  for (int k0 = 0; k0 < 256; k0 += 32) {
    v16h a0 = load_a_f16v(e1h, 256, row0 + l, h, k0);
    v16h a1 = load_a_f16v(e1h, 256, row0 + 16 + l, h, k0);
#pragma unroll
    for (int j = 0; j < 4; ++j) {
      v16h b = load_b_f16T(We2T, 256, k0, colb + 16 * j + l, h);
      acc[0][j] = wmma_f16(a0, b, acc[0][j]);
      acc[1][j] = wmma_f16(a1, b, acc[1][j]);
    }
  }
  {
    v16h a0 = load_a_g16v(GFm, eg[row0 + l], h);
    v16h a1 = load_a_g16v(GFm, eg[row0 + 16 + l], h);
#pragma unroll
    for (int j = 0; j < 4; ++j) {
      v16h b = load_b_f16T(Weg2T, 32, 0, colb + 16 * j + l, h);
      acc[0][j] = wmma_f16(a0, b, acc[0][j]);
      acc[1][j] = wmma_f16(a1, b, acc[1][j]);
    }
  }
  float bias[4];
#pragma unroll
  for (int j = 0; j < 4; ++j) bias[j] = be2[colb + 16 * j + l];
#pragma unroll
  for (int i = 0; i < 2; ++i) {
#pragma unroll
    for (int r = 0; r < 8; ++r) {
      int rd = row0 + 16 * i + 8 * h + r;
      size_t nbase = (size_t)recv[rd] * 128;
      size_t gbase = (size_t)eg[rd] * 128;
#pragma unroll
      for (int j = 0; j < 4; ++j) {
        int col = colb + 16 * j + l;
        float v = acc[i][j][r] + bias[j];
        v = v > 0.f ? v : 0.f;
        atomicAdd(&nsum2[nbase + col], v);
        atomicAdd(&gsum_e[gbase + col], v);
      }
    }
  }
}

// -------- stage 4: n2 = relu(n1@Wn2 + mean2@Win2 + G[ng]@Wng2 + bn2); -------
__global__ void k_node2(const _Float16* __restrict__ n1h, const float* __restrict__ nsum2,
                        const float* __restrict__ cnt_node, const float* __restrict__ GFm,
                        const int* __restrict__ ng, const _Float16* __restrict__ Wn2T,
                        const _Float16* __restrict__ Win2T,
                        const _Float16* __restrict__ Wng2T,
                        const float* __restrict__ bn2, float* __restrict__ gsum_n) {
  int wv = blockIdx.x * 8 + (threadIdx.x >> 5);
  int lane = threadIdx.x & 31, h = lane >> 4, l = lane & 15;
  int mp = wv >> 1, ngp = wv & 1;                 // 512 x 2
  int row0 = mp * 32, colb = ngp * 64;
  v8f acc[2][4] = {};
#pragma unroll
  for (int k0 = 0; k0 < 256; k0 += 32) {
    v16h a0 = load_a_f16v(n1h, 256, row0 + l, h, k0);
    v16h a1 = load_a_f16v(n1h, 256, row0 + 16 + l, h, k0);
#pragma unroll
    for (int j = 0; j < 4; ++j) {
      v16h b = load_b_f16T(Wn2T, 256, k0, colb + 16 * j + l, h);
      acc[0][j] = wmma_f16(a0, b, acc[0][j]);
      acc[1][j] = wmma_f16(a1, b, acc[1][j]);
    }
  }
  float ri0 = 1.0f / fmaxf(cnt_node[row0 + l], 1.0f);
  float ri1 = 1.0f / fmaxf(cnt_node[row0 + 16 + l], 1.0f);
#pragma unroll
  for (int k0 = 0; k0 < 128; k0 += 32) {
    v16h a0 = load_a_f32sv(nsum2, 128, row0 + l, h, k0, ri0);
    v16h a1 = load_a_f32sv(nsum2, 128, row0 + 16 + l, h, k0, ri1);
#pragma unroll
    for (int j = 0; j < 4; ++j) {
      v16h b = load_b_f16T(Win2T, 128, k0, colb + 16 * j + l, h);
      acc[0][j] = wmma_f16(a0, b, acc[0][j]);
      acc[1][j] = wmma_f16(a1, b, acc[1][j]);
    }
  }
  {
    v16h a0 = load_a_g16v(GFm, ng[row0 + l], h);
    v16h a1 = load_a_g16v(GFm, ng[row0 + 16 + l], h);
#pragma unroll
    for (int j = 0; j < 4; ++j) {
      v16h b = load_b_f16T(Wng2T, 32, 0, colb + 16 * j + l, h);
      acc[0][j] = wmma_f16(a0, b, acc[0][j]);
      acc[1][j] = wmma_f16(a1, b, acc[1][j]);
    }
  }
  float bias[4];
#pragma unroll
  for (int j = 0; j < 4; ++j) bias[j] = bn2[colb + 16 * j + l];
#pragma unroll
  for (int i = 0; i < 2; ++i) {
#pragma unroll
    for (int r = 0; r < 8; ++r) {
      int rd = row0 + 16 * i + 8 * h + r;
      size_t gbase = (size_t)ng[rd] * 128;
#pragma unroll
      for (int j = 0; j < 4; ++j) {
        int col = colb + 16 * j + l;
        float v = acc[i][j][r] + bias[j];
        v = v > 0.f ? v : 0.f;
        atomicAdd(&gsum_n[gbase + col], v);
      }
    }
  }
}

// -------- stage 5: g = relu(nbar@Ugn + ebar@Uge + G@Ugg + bg) ---------------
__global__ void k_global(const float* __restrict__ gsum_n, const float* __restrict__ gsum_e,
                         const float* __restrict__ cnt_ng, const float* __restrict__ cnt_eg,
                         const float* __restrict__ GFm, const _Float16* __restrict__ UgnT,
                         const _Float16* __restrict__ UgeT,
                         const _Float16* __restrict__ UggT,
                         const float* __restrict__ bg, float* __restrict__ gbuf) {
  int wv = blockIdx.x * 8 + (threadIdx.x >> 5);
  int lane = threadIdx.x & 31, h = lane >> 4, l = lane & 15;
  int mp = wv >> 2, ngp = wv & 3;                 // 8 x 4
  int row0 = mp * 32, colb = ngp * 64;
  v8f acc[2][4] = {};
  float rn0 = 1.0f / fmaxf(cnt_ng[row0 + l], 1.0f);
  float rn1 = 1.0f / fmaxf(cnt_ng[row0 + 16 + l], 1.0f);
  float re0 = 1.0f / fmaxf(cnt_eg[row0 + l], 1.0f);
  float re1 = 1.0f / fmaxf(cnt_eg[row0 + 16 + l], 1.0f);
#pragma unroll
  for (int k0 = 0; k0 < 128; k0 += 32) {
    v16h a0 = load_a_f32sv(gsum_n, 128, row0 + l, h, k0, rn0);
    v16h a1 = load_a_f32sv(gsum_n, 128, row0 + 16 + l, h, k0, rn1);
#pragma unroll
    for (int j = 0; j < 4; ++j) {
      v16h b = load_b_f16T(UgnT, 128, k0, colb + 16 * j + l, h);
      acc[0][j] = wmma_f16(a0, b, acc[0][j]);
      acc[1][j] = wmma_f16(a1, b, acc[1][j]);
    }
  }
#pragma unroll
  for (int k0 = 0; k0 < 128; k0 += 32) {
    v16h a0 = load_a_f32sv(gsum_e, 128, row0 + l, h, k0, re0);
    v16h a1 = load_a_f32sv(gsum_e, 128, row0 + 16 + l, h, k0, re1);
#pragma unroll
    for (int j = 0; j < 4; ++j) {
      v16h b = load_b_f16T(UgeT, 128, k0, colb + 16 * j + l, h);
      acc[0][j] = wmma_f16(a0, b, acc[0][j]);
      acc[1][j] = wmma_f16(a1, b, acc[1][j]);
    }
  }
  {
    v16h a0 = load_a_g16v(GFm, row0 + l, h);
    v16h a1 = load_a_g16v(GFm, row0 + 16 + l, h);
#pragma unroll
    for (int j = 0; j < 4; ++j) {
      v16h b = load_b_f16T(UggT, 32, 0, colb + 16 * j + l, h);
      acc[0][j] = wmma_f16(a0, b, acc[0][j]);
      acc[1][j] = wmma_f16(a1, b, acc[1][j]);
    }
  }
  float bias[4];
#pragma unroll
  for (int j = 0; j < 4; ++j) bias[j] = bg[colb + 16 * j + l];
#pragma unroll
  for (int i = 0; i < 2; ++i) {
#pragma unroll
    for (int r = 0; r < 8; ++r) {
      int rd = row0 + 16 * i + 8 * h + r;
#pragma unroll
      for (int j = 0; j < 4; ++j) {
        float v = acc[i][j][r] + bias[j];
        gbuf[(size_t)rd * 256 + colb + 16 * j + l] = v > 0.f ? v : 0.f;
      }
    }
  }
}

// -------- stage 6: heads (256x8 each, K=256) — tiny, plain VALU -------------
__global__ void k_heads(const float* __restrict__ gbuf, const float* __restrict__ Wm,
                        const float* __restrict__ bm, const float* __restrict__ Ws,
                        const float* __restrict__ bs, float* __restrict__ out) {
  int t = blockIdx.x * blockDim.x + threadIdx.x;  // 0..4095
  int b = t >> 4, j = t & 15;
  const float* grow = gbuf + (size_t)b * 256;
  if (j < 8) {
    float s = bm[j];
    for (int k = 0; k < 256; ++k) s += grow[k] * Wm[k * 8 + j];
    out[b * 8 + j] = s;
  } else {
    int jj = j - 8;
    float s = bs[jj];
    for (int k = 0; k < 256; ++k) s += grow[k] * Ws[k * 8 + jj];
    s = fminf(fmaxf(s, -20.0f), 2.0f);
    out[2048 + b * 8 + jj] = s;
  }
}

extern "C" void kernel_launch(void* const* d_in, const int* in_sizes, int n_in,
                              void* d_out, int out_size, void* d_ws, size_t ws_size,
                              hipStream_t stream) {
  const float* node_feats   = (const float*)d_in[0];
  const float* edge_feats   = (const float*)d_in[1];
  const float* global_feats = (const float*)d_in[2];
  const int*   receivers    = (const int*)d_in[3];
  const int*   node_graph   = (const int*)d_in[4];
  const int*   edge_graph   = (const int*)d_in[5];
  const float* We1  = (const float*)d_in[6];
  const float* be1  = (const float*)d_in[7];
  const float* Wn1  = (const float*)d_in[8];
  const float* Win1 = (const float*)d_in[9];
  const float* bn1  = (const float*)d_in[10];
  const float* We2  = (const float*)d_in[11];
  const float* Weg2 = (const float*)d_in[12];
  const float* be2  = (const float*)d_in[13];
  const float* Wn2  = (const float*)d_in[14];
  const float* Win2 = (const float*)d_in[15];
  const float* Wng2 = (const float*)d_in[16];
  const float* bn2  = (const float*)d_in[17];
  const float* Ugn  = (const float*)d_in[18];
  const float* Uge  = (const float*)d_in[19];
  const float* Ugg  = (const float*)d_in[20];
  const float* bg   = (const float*)d_in[21];
  const float* Wm   = (const float*)d_in[22];
  const float* bm   = (const float*)d_in[23];
  const float* Ws   = (const float*)d_in[24];
  const float* bs   = (const float*)d_in[25];
  float* out = (float*)d_out;

  // ---- workspace layout ----
  char* ws = (char*)d_ws;
  float* nsum1    = (float*)ws;                           // 16384*256
  float* nsum2    = nsum1 + (size_t)N_NODES * 256;        // 16384*128
  float* gsum_n   = nsum2 + (size_t)N_NODES * 128;        // 256*128
  float* gsum_e   = gsum_n + (size_t)NGRAPH * 128;        // 256*128
  float* cnt_node = gsum_e + (size_t)NGRAPH * 128;        // 16384
  float* cnt_eg   = cnt_node + N_NODES;                   // 256
  float* cnt_ng   = cnt_eg + NGRAPH;                      // 256
  size_t accN = (size_t)(cnt_ng + NGRAPH - nsum1);        // floats to zero
  _Float16* e1h = (_Float16*)(cnt_ng + NGRAPH);           // 131072*256 f16
  _Float16* n1h = e1h + (size_t)N_EDGES * 256;            // 16384*256 f16
  float* gbuf   = (float*)(n1h + (size_t)N_NODES * 256);  // 256*256 f32
  // pre-transposed f16 weights, K zero-padded to Kp
  _Float16* wt = (_Float16*)(gbuf + (size_t)NGRAPH * 256);
  _Float16* We1T  = wt;                 // 256 x 32
  _Float16* Wn1T  = We1T  + 256 * 32;   // 256 x 64
  _Float16* Win1T = Wn1T  + 256 * 64;   // 256 x 256
  _Float16* We2T  = Win1T + 256 * 256;  // 128 x 256
  _Float16* Weg2T = We2T  + 128 * 256;  // 128 x 32 (K=16 padded)
  _Float16* Wn2T  = Weg2T + 128 * 32;   // 128 x 256
  _Float16* Win2T = Wn2T  + 128 * 256;  // 128 x 128
  _Float16* Wng2T = Win2T + 128 * 128;  // 128 x 32 (K=16 padded)
  _Float16* UgnT  = Wng2T + 128 * 32;   // 256 x 128
  _Float16* UgeT  = UgnT  + 256 * 128;  // 256 x 128
  _Float16* UggT  = UgeT  + 256 * 128;  // 256 x 32 (K=16 padded)

  // zero accumulators
  k_zero<<<(int)((accN + 255) / 256), 256, 0, stream>>>(nsum1, (int)accN);
  // weight prep: transpose + f16 convert (K, N, Kp)
  k_wprep<<<(256 * 32 + 255) / 256, 256, 0, stream>>>(We1,  We1T,  32,  256, 32);
  k_wprep<<<(256 * 64 + 255) / 256, 256, 0, stream>>>(Wn1,  Wn1T,  64,  256, 64);
  k_wprep<<<(256 * 256 + 255) / 256, 256, 0, stream>>>(Win1, Win1T, 256, 256, 256);
  k_wprep<<<(128 * 256 + 255) / 256, 256, 0, stream>>>(We2,  We2T,  256, 128, 256);
  k_wprep<<<(128 * 32 + 255) / 256, 256, 0, stream>>>(Weg2, Weg2T, 16,  128, 32);
  k_wprep<<<(128 * 256 + 255) / 256, 256, 0, stream>>>(Wn2,  Wn2T,  256, 128, 256);
  k_wprep<<<(128 * 128 + 255) / 256, 256, 0, stream>>>(Win2, Win2T, 128, 128, 128);
  k_wprep<<<(128 * 32 + 255) / 256, 256, 0, stream>>>(Wng2, Wng2T, 16,  128, 32);
  k_wprep<<<(256 * 128 + 255) / 256, 256, 0, stream>>>(Ugn,  UgnT,  128, 256, 128);
  k_wprep<<<(256 * 128 + 255) / 256, 256, 0, stream>>>(Uge,  UgeT,  128, 256, 128);
  k_wprep<<<(256 * 32 + 255) / 256, 256, 0, stream>>>(Ugg,  UggT,  16,  256, 32);
  // counts (segment-mean divisors)
  k_count_edges<<<N_EDGES / 256, 256, 0, stream>>>(receivers, edge_graph, cnt_node, cnt_eg);
  k_count_nodes<<<N_NODES / 256, 256, 0, stream>>>(node_graph, cnt_ng);
  // edge block 1: 4096 row-pairs x 4 col-groups = 16384 waves
  k_edge1<<<2048, 256, 0, stream>>>(edge_feats, We1T, be1, receivers, e1h, nsum1);
  // node block 1: 512 x 4 = 2048 waves
  k_node1<<<256, 256, 0, stream>>>(node_feats, nsum1, cnt_node, Wn1T, Win1T, bn1, n1h);
  // edge block 2: 4096 x 2 = 8192 waves, fused scatter of e2
  k_edge2<<<1024, 256, 0, stream>>>(e1h, global_feats, edge_graph, receivers,
                                    We2T, Weg2T, be2, nsum2, gsum_e);
  // node block 2: 512 x 2 = 1024 waves, fused scatter of n2
  k_node2<<<128, 256, 0, stream>>>(n1h, nsum2, cnt_node, global_feats, node_graph,
                                   Wn2T, Win2T, Wng2T, bn2, gsum_n);
  // global block: 8 x 4 = 32 waves
  k_global<<<4, 256, 0, stream>>>(gsum_n, gsum_e, cnt_ng, cnt_eg, global_feats,
                                  UgnT, UgeT, UggT, bg, gbuf);
  // heads
  k_heads<<<16, 256, 0, stream>>>(gbuf, Wm, bm, Ws, bs, out);
}
